// ThreeLayerNet_1LevelAttn_multihead_54546084660143
// MI455X (gfx1250) — compile-verified, matrix operand order checked by reading, not run on hardware
//
#include <hip/hip_runtime.h>
#include <math.h>

// ---- CDNA5 (gfx1250) wave32 WMMA types ----
typedef __attribute__((ext_vector_type(16))) __bf16 v16bf;
typedef __attribute__((ext_vector_type(8)))  float  v8f;
typedef __attribute__((ext_vector_type(4)))  __bf16 v4bf;
typedef __attribute__((ext_vector_type(4)))  unsigned int u32x4;
typedef __attribute__((ext_vector_type(8)))  int i32x8;
typedef __attribute__((ext_vector_type(4)))  int i32x4;

#if defined(__has_builtin)
#  if __has_builtin(__builtin_amdgcn_tensor_load_to_lds)
#    define HAVE_TDM 1
#  endif
#endif
#ifndef HAVE_TDM
#  define HAVE_TDM 0
#endif

#define B_  16
#define U_  12
#define V_  2048
#define D_  256
#define NH  4
#define H1_ 1024
#define H2_ 512

// attention kernel tiling
#define VT      256                    // V rows staged per chunk
#define NCHUNK  (V_ / VT)              // 8
#define LSTRIDE 264                    // padded bf16 row stride: 528B (16B aligned, banks spread)

// dynamic LDS layout (bytes)
#define SM_XB    0
#define SM_WL    (VT * LSTRIDE * 2)              // 135168
#define SM_SC    (SM_WL + 256 * LSTRIDE * 2)     // 270336
#define SM_SUME  (SM_SC + VT * NH * 4)           // 274432
#define SM_TOTAL (SM_SUME + 16)                  // 274448  (<= 320KB WGP LDS)

// Load a 16x32 bf16 WMMA fragment (A rows of X / B cols of W^T == rows of W).
// Per ISA layout: lanes 0-15 hold K {e0..e0+7, e1..e1+7} with e0=kb, e1=kb+16;
// lanes 16-31 the same with +8.  Two ds/global b128 loads.
__device__ __forceinline__ v16bf frag16(const __bf16* p, int e0, int e1) {
  union { uint4 q[2]; v16bf v; } u;
  u.q[0] = *(const uint4*)(p + e0);
  u.q[1] = *(const uint4*)(p + e1);
  return u.v;
}

// Same fragment but sourced from fp32 memory with on-the-fly bf16 convert.
__device__ __forceinline__ v16bf frag16f(const float* p, int e0, int e1) {
  union { v16bf v; __bf16 e[16]; } u;
#pragma unroll
  for (int i = 0; i < 8; ++i) u.e[i]     = (__bf16)p[e0 + i];
#pragma unroll
  for (int i = 0; i < 8; ++i) u.e[8 + i] = (__bf16)p[e1 + i];
  return u.v;
}

__device__ __forceinline__ v8f wmma_bf16(v16bf a, v16bf b, v8f c) {
  return __builtin_amdgcn_wmma_f32_16x16x32_bf16(false, a, false, b, (short)0, c,
                                                 false, false);
}

// ---------------- kernel 0: W1..W4 (fp32) -> Wb (bf16) ----------------
__global__ void k0_convw(const float* __restrict__ W1, const float* __restrict__ W2,
                         const float* __restrict__ W3, const float* __restrict__ W4,
                         __bf16* __restrict__ Wb) {
  int gid = blockIdx.x * blockDim.x + threadIdx.x;   // 4*65536 threads
  int h = gid >> 16, r = gid & 65535;
  const float* src = (h == 0) ? W1 : (h == 1) ? W2 : (h == 2) ? W3 : W4;
  Wb[gid] = (__bf16)src[r];
}

// ---------------- kernel 1: fused bilinear attention ----------------
// grid = B*U (192), block = 256 (8 waves). One (b,u) per workgroup.
// Per 256-row V chunk: stage X tile (bf16) + per-head W (bf16) in LDS (TDM),
// WMMA T = X @ W^T, fuse S = rowsum(T .* X), exp((tanh(S)+1)*mask),
// accumulate unnormalized pooled sum; normalize at the end -> Xf (bf16).
__global__ void __launch_bounds__(256, 1)
k1_attn(const float* __restrict__ X, const float* __restrict__ maskP,
        const __bf16* __restrict__ Wb, __bf16* __restrict__ Xf) {
  extern __shared__ char smem[];
  __bf16* Xb   = (__bf16*)(smem + SM_XB);   // [VT][LSTRIDE]
  __bf16* Wl   = (__bf16*)(smem + SM_WL);   // [256][LSTRIDE]
  float*  Sc   = (float*)(smem + SM_SC);    // [VT][NH] scores -> exp weights
  float*  sumE = (float*)(smem + SM_SUME);  // [NH] softmax denominators

  const int bu = blockIdx.x;
  const int b  = bu / U_, u = bu % U_;
  const float* Xbu = X + (size_t)bu * V_ * D_;
  const float* Mbu = maskP + (size_t)bu * V_;

  const int t     = threadIdx.x;
  const int lane  = t & 31;
  const int w     = t >> 5;
  const int laneN = lane & 15;
  const int half  = lane >> 4;

  if (t < NH) sumE[t] = 0.0f;
  float acc[NH] = {0.f, 0.f, 0.f, 0.f};   // pooled sums for column d = t

  for (int ck = 0; ck < NCHUNK; ++ck) {
    const float* Xck = Xbu + (size_t)ck * VT * D_;
    if (ck + 1 < NCHUNK)  // hint next chunk (1KB per thread covers 256KB)
      __builtin_prefetch((const char*)(Xck + VT * D_) + t * 1024, 0, 0);

    __syncthreads();      // previous pooling done before Xb overwrite
    // ---- stage X chunk fp32 -> bf16 LDS ----
    for (int i = t; i < VT * D_ / 4; i += 256) {
      int row = i >> 6, col = (i & 63) * 4;
      float4 f = *(const float4*)(Xck + row * D_ + col);
      v4bf o; o.x = (__bf16)f.x; o.y = (__bf16)f.y; o.z = (__bf16)f.z; o.w = (__bf16)f.w;
      *(v4bf*)(Xb + row * LSTRIDE + col) = o;
    }

    for (int h = 0; h < NH; ++h) {
      __syncthreads();    // Xb ready / previous head's compute done
      // ---- stage W_h (bf16, L2-resident) into LDS ----
      const __bf16* WbH = Wb + h * (D_ * D_);
#if HAVE_TDM
      // Tensor Data Mover: one DMA moves the 256x256 bf16 tile; pad_amount=4
      // DWORDs every pad_interval=128 DWORDs reproduces the 264-elem row stride.
      if (w == 0) {
        unsigned long long ga = (unsigned long long)(uintptr_t)WbH;
        unsigned int ldsoff = (unsigned int)(uintptr_t)(void*)Wl;  // LDS = addr[31:0]
        u32x4 g0 = { 1u,                                   // count=1 (valid D#)
                     ldsoff,                               // lds_addr
                     (unsigned int)ga,                     // global_addr[31:0]
                     (unsigned int)((ga >> 32) & 0x01FFFFFFu) | 0x80000000u }; // [56:32] | type=2
        i32x8 g1 = { (int)((1u << 16) |                    // data_size = 2 bytes
                           (1u << 20) |                    // pad_enable
                           (6u << 22) |                    // pad_interval: 128 DWORDs
                           (3u << 25)),                    // pad_amount: 4 DWORDs
                     (int)(256u << 16),                    // tensor_dim0 = 256 (lo16)
                     (int)(256u << 16),                    // dim0 hi=0 | tensor_dim1 lo16
                     (int)(256u << 16),                    // dim1 hi=0 | tile_dim0 = 256
                     (int)256,                             // tile_dim1 = 256, tile_dim2 = 0
                     (int)256,                             // tensor_dim0_stride = 256
                     0, 0 };                               // dim0_stride hi, dim1_stride
        i32x4 g2 = { 0, 0, 0, 0 };
        i32x4 g3 = { 0, 0, 0, 0 };
#if __clang_major__ >= 23
        i32x8 g4 = { 0, 0, 0, 0, 0, 0, 0, 0 };
        __builtin_amdgcn_tensor_load_to_lds(g0, g1, g2, g3, g4, 0);
#else
        __builtin_amdgcn_tensor_load_to_lds(g0, g1, g2, g3, 0);
#endif
#if __has_builtin(__builtin_amdgcn_s_wait_tensorcnt)
        __builtin_amdgcn_s_wait_tensorcnt(0);
#else
        asm volatile("s_wait_tensorcnt 0x0" ::: "memory");
#endif
      }
#else
      for (int i = t; i < D_ * D_ / 8; i += 256) {
        int row = i >> 5, col = (i & 31) * 8;
        *(uint4*)(Wl + row * LSTRIDE + col) = *(const uint4*)(WbH + row * D_ + col);
      }
#endif
      __syncthreads();

#pragma unroll 1
      for (int mt = 0; mt < 2; ++mt) {
        const int r0 = (w * 2 + mt) * 16;
        // A fragments: rows of X tile (K = 256 -> 8 k-steps of 32)
        v16bf a[8];
#pragma unroll
        for (int kt = 0; kt < 8; ++kt) {
          int kb = kt * 32 + half * 8;
          a[kt] = frag16(Xb + (r0 + laneN) * LSTRIDE, kb, kb + 16);
        }
        float pr[8] = {0.f, 0.f, 0.f, 0.f, 0.f, 0.f, 0.f, 0.f};
#pragma unroll 1
        for (int nt = 0; nt < 16; ++nt) {
          v8f c = {};
          const __bf16* wrow = Wl + (nt * 16 + laneN) * LSTRIDE;
#pragma unroll
          for (int kt = 0; kt < 8; ++kt) {
            int kb = kt * 32 + half * 8;
            c = wmma_bf16(a[kt], frag16(wrow, kb, kb + 16), c);
          }
          // fuse S += T[.,n] * X[.,n]  (C layout: vgpr r -> M=half*8+r, N=laneN)
          const __bf16* xcol = Xb + (r0 + half * 8) * LSTRIDE + nt * 16 + laneN;
#pragma unroll
          for (int r = 0; r < 8; ++r)
            pr[r] += c[r] * (float)xcol[r * LSTRIDE];
        }
        // reduce over N (16-lane groups), write scores
#pragma unroll
        for (int r = 0; r < 8; ++r) {
          float s = pr[r];
          s += __shfl_xor(s, 1, 32);
          s += __shfl_xor(s, 2, 32);
          s += __shfl_xor(s, 4, 32);
          s += __shfl_xor(s, 8, 32);
          if (laneN == r) Sc[(r0 + half * 8 + r) * NH + h] = s;
        }
      }
      __syncthreads();
      // ---- e = exp((tanh(S)+1)*mask); logits bounded in [0,2] -> no max needed ----
      {
        float sv = Sc[t * NH + h];
        float m  = Mbu[ck * VT + t];
        float e  = __expf((tanhf(sv) + 1.0f) * m);
        Sc[t * NH + h] = e;
        float es = e;
        es += __shfl_xor(es, 1, 32);
        es += __shfl_xor(es, 2, 32);
        es += __shfl_xor(es, 4, 32);
        es += __shfl_xor(es, 8, 32);
        es += __shfl_xor(es, 16, 32);
        if (lane == 0) atomicAdd(&sumE[h], es);   // ds_add_f32
      }
    }
    __syncthreads();
    // ---- fused pooling: acc[h] += e[h][v] * X[v][d], d = t ----
#pragma unroll 1
    for (int v = 0; v < VT; ++v) {
      float4 e4 = *(const float4*)(Sc + v * NH);     // LDS broadcast
      float  xv = (float)Xb[v * LSTRIDE + t];
      acc[0] += e4.x * xv; acc[1] += e4.y * xv;
      acc[2] += e4.z * xv; acc[3] += e4.w * xv;
    }
  }
  __syncthreads();
#pragma unroll
  for (int h = 0; h < NH; ++h) {
    float val = acc[h] / sumE[h];
    Xf[(size_t)b * (U_ * NH * D_) + (u * NH + h) * D_ + t] = (__bf16)val;
  }
}

// ---------------- kernel 2: h1 = relu(Xf @ L1w^T + b1) ----------------
// grid = 64 (N tiles of 16), block = 256 (8 waves split K = 12288).
__global__ void __launch_bounds__(256)
k2_l1(const __bf16* __restrict__ Xf, const float* __restrict__ L1w,
      const float* __restrict__ L1b, __bf16* __restrict__ h1) {
  __shared__ float red[8 * 256];
  const int t = threadIdx.x, lane = t & 31, w = t >> 5;
  const int laneN = lane & 15, half = lane >> 4;
  const int j0 = blockIdx.x * 16;
  const int k0 = w * 1536;                          // K slice per wave (48 steps)

  const __bf16* arow = Xf + (size_t)laneN * (U_ * NH * D_);
  const float*  brow = L1w + (size_t)(j0 + laneN) * (U_ * NH * D_);
  v8f c = {};
#pragma unroll 1
  for (int kt = 0; kt < 48; ++kt) {
    int kb = k0 + kt * 32 + half * 8;
    c = wmma_bf16(frag16(arow, kb, kb + 16), frag16f(brow, kb, kb + 16), c);
  }
#pragma unroll
  for (int r = 0; r < 8; ++r)
    red[w * 256 + (half * 8 + r) * 16 + laneN] = c[r];
  __syncthreads();
  const int m = t >> 4, n = t & 15;
  float s = 0.f;
#pragma unroll
  for (int ww = 0; ww < 8; ++ww) s += red[ww * 256 + m * 16 + n];
  s = fmaxf(s + L1b[j0 + n], 0.0f);
  h1[(size_t)m * H1_ + j0 + n] = (__bf16)s;
}

// ---------------- kernel 3: h2 = relu(h1 @ L2w^T + b2) ----------------
// grid = 32 (N tiles), block = 32 (one wave, full K = 1024).
__global__ void k3_l2(const __bf16* __restrict__ h1, const float* __restrict__ L2w,
                      const float* __restrict__ L2b, float* __restrict__ h2) {
  const int lane = threadIdx.x & 31, laneN = lane & 15, half = lane >> 4;
  const int j0 = blockIdx.x * 16;
  const __bf16* arow = h1 + (size_t)laneN * H1_;
  const float*  brow = L2w + (size_t)(j0 + laneN) * H1_;
  v8f c = {};
#pragma unroll 1
  for (int kt = 0; kt < 32; ++kt) {
    int kb = kt * 32 + half * 8;
    c = wmma_bf16(frag16(arow, kb, kb + 16), frag16f(brow, kb, kb + 16), c);
  }
#pragma unroll
  for (int r = 0; r < 8; ++r) {
    int m = half * 8 + r, n = j0 + laneN;
    h2[(size_t)m * H2_ + n] = fmaxf(c[r] + L2b[n], 0.0f);
  }
}

// ---------------- kernel 4: out = h2 @ L3w^T + b3  (16x2) ----------------
__global__ void k4_l3(const float* __restrict__ h2, const float* __restrict__ L3w,
                      const float* __restrict__ L3b, float* __restrict__ out) {
  const int t = threadIdx.x;        // 32 lanes: m = t>>1, y = t&1
  const int m = t >> 1, y = t & 1;
  float acc = L3b[y];
#pragma unroll 4
  for (int k = 0; k < H2_; ++k) acc += h2[m * H2_ + k] * L3w[y * H2_ + k];
  out[m * 2 + y] = acc;
}

extern "C" void kernel_launch(void* const* d_in, const int* in_sizes, int n_in,
                              void* d_out, int out_size, void* d_ws, size_t ws_size,
                              hipStream_t stream) {
  (void)in_sizes; (void)n_in; (void)out_size; (void)ws_size;
  const float* X    = (const float*)d_in[0];
  const float* mask = (const float*)d_in[1];
  const float* W1   = (const float*)d_in[2];
  const float* W2   = (const float*)d_in[3];
  const float* W3   = (const float*)d_in[4];
  const float* W4   = (const float*)d_in[5];
  const float* L1w  = (const float*)d_in[6];
  const float* L1b  = (const float*)d_in[7];
  const float* L2w  = (const float*)d_in[8];
  const float* L2b  = (const float*)d_in[9];
  const float* L3w  = (const float*)d_in[10];
  const float* L3b  = (const float*)d_in[11];
  float* out = (float*)d_out;

  char* ws = (char*)d_ws;
  __bf16* Wb = (__bf16*)(ws + 0);        // 4*256*256 bf16  = 512 KB
  __bf16* Xf = (__bf16*)(ws + 524288);   // 16*12288 bf16   = 384 KB
  __bf16* h1 = (__bf16*)(ws + 917504);   // 16*1024 bf16    = 32 KB
  float*  h2 = (float*)(ws + 950272);    // 16*512  f32     = 32 KB

  k0_convw<<<1024, 256, 0, stream>>>(W1, W2, W3, W4, Wb);
  k1_attn<<<B_ * U_, 256, SM_TOTAL, stream>>>(X, mask, Wb, Xf);
  k2_l1<<<H1_ / 16, 256, 0, stream>>>(Xf, L1w, L1b, h1);
  k3_l2<<<H2_ / 16, 32, 0, stream>>>(h1, L2w, L2b, h2);
  k4_l3<<<1, 32, 0, stream>>>(h2, L3w, L3b, out);
}
